// Mamba3Block_45681272160520
// MI455X (gfx1250) — matrix operand order repaired and checked by reading, not compiled, not run on hardware
//
#include <hip/hip_runtime.h>
#include <hip/hip_bf16.h>

// ---------------------------------------------------------------------------
// Mamba3-style block on MI455X (gfx1250), wave32 + WMMA f16->f32,
// async global->LDS tile staging (ASYNCcnt) with LDS double buffering.
//   B=8, L=2048, D=1024, FF_MULT=2, M = B*L = 16384 tokens.
// fwd/bwd SSM branches are position-independent MLPs (flips cancel), so they
// are fused into one [w1|bwd_w1] / [w2;bwd_w2] GEMM pair.
// ---------------------------------------------------------------------------

typedef __attribute__((ext_vector_type(16))) _Float16 v16h;
typedef __attribute__((ext_vector_type(8)))  float    v8f;

union Frag {
    v16h v;
    unsigned u[8];
};

__device__ __forceinline__ float gelu_erf_f(float x) {
    return 0.5f * x * (1.0f + erff(x * 0.70710678118654752f));
}
__device__ __forceinline__ float gelu_tanh_f(float x) {
    float x3 = x * x * x;
    return 0.5f * x * (1.0f + tanhf(0.7978845608028654f * (x + 0.044715f * x3)));
}
__device__ __forceinline__ float silu_f(float x) {
    return x / (1.0f + expf(-x));
}

// async global -> LDS copy, 16B per lane (CDNA5, tracked by ASYNCcnt).
// Per ISA 08_async_tensor: imm offset applies to both the LDS and global address.
#define ASYNC_G2L_B128(ldsoff, gptr, imm)                                     \
    asm volatile("global_load_async_to_lds_b128 %0, %1, off offset:" #imm    \
                 :: "v"(ldsoff), "v"(gptr) : "memory")

__device__ __forceinline__ void wait_async0() {
#if __has_builtin(__builtin_amdgcn_s_wait_asynccnt)
    __builtin_amdgcn_s_wait_asynccnt(0);
#else
    asm volatile("s_wait_asynccnt 0x0" ::: "memory");
#endif
}

// ---------------------------------------------------------------------------
// Small prep kernels
// ---------------------------------------------------------------------------

// bias concat: b1cat[0:1024]=ssm_b1, b1cat[1024:2048]=bwd_b1 ; b2sum=ssm_b2+bwd_b2
__global__ void bias_prep_kernel(const float* __restrict__ ssm_b1,
                                 const float* __restrict__ bwd_b1,
                                 const float* __restrict__ ssm_b2,
                                 const float* __restrict__ bwd_b2,
                                 float* __restrict__ b1cat,
                                 float* __restrict__ b2sum) {
    int i = blockIdx.x * 256 + threadIdx.x;   // 0..2047
    if (i < 1024) {
        b1cat[i] = ssm_b1[i];
        b2sum[i] = ssm_b2[i] + bwd_b2[i];
    } else if (i < 2048) {
        b1cat[i] = bwd_b1[i - 1024];
    }
}

// transpose f32 (K x N, row-major) -> f16 dst[n*ldd + k] (+ dst_off elements)
__global__ void transpose_to_f16_kernel(const float* __restrict__ src,
                                        _Float16* __restrict__ dst,
                                        int K, int N, int ldd, long long dst_off) {
    __shared__ float tile[32][33];
    int k0 = blockIdx.y * 32, n0 = blockIdx.x * 32;
    int tx = threadIdx.x, ty = threadIdx.y;    // block (32, 8)
#pragma unroll
    for (int i = 0; i < 32; i += 8)
        tile[ty + i][tx] = src[(size_t)(k0 + ty + i) * N + (n0 + tx)];
    __syncthreads();
#pragma unroll
    for (int i = 0; i < 32; i += 8) {
        int n = n0 + ty + i;
        int k = k0 + tx;
        dst[dst_off + (size_t)n * ldd + k] = (_Float16)tile[tx][ty + i];
    }
}

// adaLN modulation: out[b, n] = silu(t[b,:]) . w[:, n] + bias[n], out is 8 x 3072
__global__ void adaln_mod_kernel(const float* __restrict__ t,
                                 const float* __restrict__ w,
                                 const float* __restrict__ bias,
                                 float* __restrict__ out) {
    int idx = blockIdx.x * 256 + threadIdx.x;   // 0 .. 8*3072-1
    int b = idx / 3072;
    int n = idx - b * 3072;
    const float* tr = t + (size_t)b * 1024;
    float acc = 0.0f;
    for (int k = 0; k < 1024; ++k)
        acc = fmaf(silu_f(tr[k]), w[(size_t)k * 3072 + n], acc);
    out[idx] = acc + bias[n];
}

// LayerNorm + AdaLN modulate (+ optional mask), f32 in -> f16 out. One row/block.
__global__ __launch_bounds__(256) void ln_mod_kernel(const float* __restrict__ x,
                                                     const float* __restrict__ ada,  // B x 3072
                                                     const unsigned char* __restrict__ mask,
                                                     _Float16* __restrict__ out,
                                                     int use_mask) {
    const int row = blockIdx.x;        // 0..16383
    const int b   = row >> 11;         // row / 2048
    const int tid = threadIdx.x;
    const float* xr = x + (size_t)row * 1024;

    float v[4], s = 0.0f, s2 = 0.0f;
#pragma unroll
    for (int i = 0; i < 4; ++i) {
        float xv = xr[tid + i * 256];
        v[i] = xv;
        s += xv;
        s2 += xv * xv;
    }
    __shared__ float rs[256], rs2[256];
    rs[tid] = s; rs2[tid] = s2;
    __syncthreads();
    for (int off = 128; off > 0; off >>= 1) {
        if (tid < off) { rs[tid] += rs[tid + off]; rs2[tid] += rs2[tid + off]; }
        __syncthreads();
    }
    const float mu   = rs[0] * (1.0f / 1024.0f);
    const float var  = rs2[0] * (1.0f / 1024.0f) - mu * mu;
    const float rinv = rsqrtf(var + 1e-6f);
    const float mfac = use_mask ? (mask[row] ? 1.0f : 0.0f) : 1.0f;

    const float* adab = ada + (size_t)b * 3072;
#pragma unroll
    for (int i = 0; i < 4; ++i) {
        int d = tid + i * 256;
        float sc = 1.0f + adab[1024 + d];
        float sh = adab[d];
        out[(size_t)row * 1024 + d] = (_Float16)((((v[i] - mu) * rinv) * sc + sh) * mfac);
    }
}

// ---------------------------------------------------------------------------
// WMMA GEMM: C(MxN) = A(MxK, f16 row-major) * Bt(NxK, f16 row-major)^T
// Workgroup tile 128x256, BK=32, 8 waves as 2(M)x4(N); each wave owns a
// 64x64 sub-tile = 4x4 16x16 fragments -> 16 v_wmma per K-step.
// Tiles staged with global_load_async_to_lds_b128 into double-buffered LDS,
// prefetched one K-step ahead (ASYNCcnt overlap with WMMA).
// Epilogues:
//   0: f16 out = gelu_erf (acc + bias[n])                       (SSM hidden)
//   1: f16 out = gelu_tanh(acc + bias[n])                       (FF hidden)
//   2: f32 out = resid + gate*(mask ? acc+bias : 0)             (SSM residual)
//   3: f32 out = resid + gate*(acc+bias)                        (FF residual)
// ---------------------------------------------------------------------------
template <int EPI>
__global__ __launch_bounds__(256) void gemm_wmma_kernel(
    const _Float16* __restrict__ A,
    const _Float16* __restrict__ Bt,
    int K, int N,
    const float* __restrict__ bias,
    const float* __restrict__ resid,
    const float* __restrict__ ada,            // B x 3072, gate at col 2048+n
    const unsigned char* __restrict__ mask,
    _Float16* __restrict__ outH,
    float* __restrict__ outF) {

    __shared__ __align__(16) _Float16 Atile[2][128][32];
    __shared__ __align__(16) _Float16 Btile[2][256][32];

    const int t  = threadIdx.x;
    const int m0 = blockIdx.y * 128;
    const int n0 = blockIdx.x * 256;

    const int lane   = t & 31;
    const int wid    = t >> 5;
    const int wave_m = wid & 1;   // 2 waves along M (64 rows each)
    const int wave_n = wid >> 1;  // 4 waves along N (64 cols each)
    const int half   = lane >> 4;
    const int l16    = lane & 15;

    v8f zero = {0.f, 0.f, 0.f, 0.f, 0.f, 0.f, 0.f, 0.f};
    v8f acc[4][4];
#pragma unroll
    for (int mi = 0; mi < 4; ++mi)
#pragma unroll
        for (int ni = 0; ni < 4; ++ni)
            acc[mi][ni] = zero;

    // staging assignments: A tile 128x32 (2 x b128/thread), B tile 256x32 (4 x b128/thread)
    const int arow = t >> 1;           // 0..127
    const int acol = (t & 1) << 4;     // 0 or 16 (f16 elements)
    const int brow = t;                // 0..255
    const char* gA = (const char*)(A  + (size_t)(m0 + arow) * K + acol);
    const char* gB = (const char*)(Bt + (size_t)(n0 + brow) * K);
    unsigned lA[2], lB[2];
    lA[0] = (unsigned)(size_t)&Atile[0][arow][acol];
    lA[1] = (unsigned)(size_t)&Atile[1][arow][acol];
    lB[0] = (unsigned)(size_t)&Btile[0][brow][0];
    lB[1] = (unsigned)(size_t)&Btile[1][brow][0];

    const int nk = K >> 5;             // K / 32

    // prologue: fill buffer 0
    {
        ASYNC_G2L_B128(lA[0], gA, 0);
        ASYNC_G2L_B128(lA[0], gA, 16);
        ASYNC_G2L_B128(lB[0], gB, 0);
        ASYNC_G2L_B128(lB[0], gB, 16);
        ASYNC_G2L_B128(lB[0], gB, 32);
        ASYNC_G2L_B128(lB[0], gB, 48);
    }

    for (int i = 0; i < nk; ++i) {
        wait_async0();                 // this wave's async tile writes landed
        __syncthreads();               // all waves' writes landed / prev reads done

        if (i + 1 < nk) {              // prefetch next K-step into other buffer
            const int nb = (i + 1) & 1;
            const char* a = gA + (size_t)(i + 1) * 64;   // 32 f16 = 64 B
            const char* b = gB + (size_t)(i + 1) * 64;
            ASYNC_G2L_B128(lA[nb], a, 0);
            ASYNC_G2L_B128(lA[nb], a, 16);
            ASYNC_G2L_B128(lB[nb], b, 0);
            ASYNC_G2L_B128(lB[nb], b, 16);
            ASYNC_G2L_B128(lB[nb], b, 32);
            ASYNC_G2L_B128(lB[nb], b, 48);
        }

        const int cur = i & 1;
        // fragment packing per CDNA5 ISA 7.12.2 VGPR layouts
        Frag af[4], bf[4];
#pragma unroll
        for (int mi = 0; mi < 4; ++mi) {
            int r = wave_m * 64 + mi * 16 + l16;
#pragma unroll
            for (int j = 0; j < 8; ++j) {
                int k0 = ((j >> 2) << 4) + ((j & 3) << 1) + (half << 3);
                af[mi].u[j] = *(const unsigned*)&Atile[cur][r][k0];
            }
        }
#pragma unroll
        for (int ni = 0; ni < 4; ++ni) {
            int c = wave_n * 64 + ni * 16 + l16;
#pragma unroll
            for (int j = 0; j < 8; ++j) {
                int k0 = ((j >> 2) << 4) + ((j & 3) << 1) + (half << 3);
                bf[ni].u[j] = *(const unsigned*)&Btile[cur][c][k0];
            }
        }
#pragma unroll
        for (int mi = 0; mi < 4; ++mi)
#pragma unroll
            for (int ni = 0; ni < 4; ++ni)
                acc[mi][ni] = __builtin_amdgcn_wmma_f32_16x16x32_f16(
                    false, af[mi].v, false, bf[ni].v,
                    (short)0, acc[mi][ni], false, false);
    }

    // epilogue: C lane mapping (m = j + 8*half, n = l16)
#pragma unroll
    for (int mi = 0; mi < 4; ++mi) {
#pragma unroll
        for (int ni = 0; ni < 4; ++ni) {
            int gn = n0 + wave_n * 64 + ni * 16 + l16;
#pragma unroll
            for (int j = 0; j < 8; ++j) {
                int gm = m0 + wave_m * 64 + mi * 16 + j + half * 8;
                float v = acc[mi][ni][j] + bias[gn];
                if (EPI == 0) {
                    outH[(size_t)gm * N + gn] = (_Float16)gelu_erf_f(v);
                } else if (EPI == 1) {
                    outH[(size_t)gm * N + gn] = (_Float16)gelu_tanh_f(v);
                } else {
                    int b = gm >> 11;                         // gm / 2048
                    float gate = ada[(size_t)b * 3072 + 2048 + gn];
                    if (EPI == 2) v = mask[gm] ? v : 0.0f;
                    size_t o = (size_t)gm * 1024 + gn;
                    outF[o] = resid[o] + gate * v;
                }
            }
        }
    }
}

// ---------------------------------------------------------------------------
// host-side launcher
// ---------------------------------------------------------------------------
extern "C" void kernel_launch(void* const* d_in, const int* in_sizes, int n_in,
                              void* d_out, int out_size, void* d_ws, size_t ws_size,
                              hipStream_t stream) {
    (void)in_sizes; (void)n_in; (void)out_size; (void)ws_size;

    constexpr int Bz = 8, L = 2048, D = 1024;
    constexpr int M  = Bz * L;           // 16384
    constexpr int D2 = 2 * D;            // 2048

    const float* x       = (const float*)d_in[0];
    const float* t       = (const float*)d_in[1];
    const unsigned char* mask = (const unsigned char*)d_in[2];
    const float* ada_w   = (const float*)d_in[3];
    const float* ada_b   = (const float*)d_in[4];
    const float* ssm_w1  = (const float*)d_in[5];
    const float* ssm_b1  = (const float*)d_in[6];
    const float* ssm_w2  = (const float*)d_in[7];
    const float* ssm_b2  = (const float*)d_in[8];
    const float* bwd_w1  = (const float*)d_in[9];
    const float* bwd_b1  = (const float*)d_in[10];
    const float* bwd_w2  = (const float*)d_in[11];
    const float* bwd_b2  = (const float*)d_in[12];
    const float* ffada_w = (const float*)d_in[13];
    const float* ffada_b = (const float*)d_in[14];
    const float* ff_w1   = (const float*)d_in[15];
    const float* ff_b1   = (const float*)d_in[16];
    const float* ff_w2   = (const float*)d_in[17];
    const float* ff_b2   = (const float*)d_in[18];
    float* out = (float*)d_out;

    // workspace carve-up
    char* ws = (char*)d_ws;
    size_t off = 0;
    auto alloc = [&](size_t bytes) -> void* {
        void* p = ws + off;
        off += (bytes + 255) & ~(size_t)255;
        return p;
    };
    _Float16* normH   = (_Float16*)alloc((size_t)M * D  * 2);  // 32 MB
    _Float16* ffnormH = (_Float16*)alloc((size_t)M * D  * 2);  // 32 MB
    _Float16* H       = (_Float16*)alloc((size_t)M * D2 * 2);  // 64 MB (reused)
    float*    x1      = (float*)   alloc((size_t)M * D  * 4);  // 64 MB
    _Float16* w1t     = (_Float16*)alloc((size_t)D2 * D * 2);  // [ssm_w1|bwd_w1]^T : 2048 x 1024
    _Float16* w2t     = (_Float16*)alloc((size_t)D * D2 * 2);  // [ssm_w2;bwd_w2]^T : 1024 x 2048
    _Float16* fw1t    = (_Float16*)alloc((size_t)D2 * D * 2);  // ff_w1^T : 2048 x 1024
    _Float16* fw2t    = (_Float16*)alloc((size_t)D * D2 * 2);  // ff_w2^T : 1024 x 2048
    float*    emb     = (float*)alloc((size_t)Bz * 3 * D * 4);
    float*    ffada   = (float*)alloc((size_t)Bz * 3 * D * 4);
    float*    b1cat   = (float*)alloc((size_t)D2 * 4);
    float*    b2sum   = (float*)alloc((size_t)D  * 4);

    // --- prep: biases, weight transposes (f32 -> f16 NxK) ---
    bias_prep_kernel<<<8, 256, 0, stream>>>(ssm_b1, bwd_b1, ssm_b2, bwd_b2, b1cat, b2sum);

    dim3 tb(32, 8);
    transpose_to_f16_kernel<<<dim3(32, 32), tb, 0, stream>>>(ssm_w1, w1t, D, D, D, 0);
    transpose_to_f16_kernel<<<dim3(32, 32), tb, 0, stream>>>(bwd_w1, w1t, D, D, D, (long long)D * D);
    transpose_to_f16_kernel<<<dim3(32, 32), tb, 0, stream>>>(ssm_w2, w2t, D, D, D2, 0);
    transpose_to_f16_kernel<<<dim3(32, 32), tb, 0, stream>>>(bwd_w2, w2t, D, D, D2, D);
    transpose_to_f16_kernel<<<dim3(64, 32), tb, 0, stream>>>(ff_w1, fw1t, D, D2, D, 0);
    transpose_to_f16_kernel<<<dim3(32, 64), tb, 0, stream>>>(ff_w2, fw2t, D2, D, D2, 0);

    // --- adaLN modulation vectors from t ---
    adaln_mod_kernel<<<(Bz * 3 * D) / 256, 256, 0, stream>>>(t, ada_w,   ada_b,   emb);
    adaln_mod_kernel<<<(Bz * 3 * D) / 256, 256, 0, stream>>>(t, ffada_w, ffada_b, ffada);

    // --- stage 1: norm = mask * (LN(x)*(1+scale)+shift) ---
    ln_mod_kernel<<<M, 256, 0, stream>>>(x, emb, mask, normH, 1);

    // --- stage 2: H = gelu_erf(norm @ [w1|bwd_w1] + b1cat)   (M x 2048) ---
    gemm_wmma_kernel<0><<<dim3(D2 / 256, M / 128), 256, 0, stream>>>(
        normH, w1t, D, D2, b1cat, nullptr, nullptr, nullptr, H, nullptr);

    // --- stage 3: x1 = x + gate_msa * mask * (H @ [w2;bwd_w2] + b2sum) ---
    gemm_wmma_kernel<2><<<dim3(D / 256, M / 128), 256, 0, stream>>>(
        H, w2t, D2, D, b2sum, x, emb, mask, nullptr, x1);

    // --- stage 4: ff_norm = LN(x1)*(1+scale_mlp)+shift_mlp ---
    ln_mod_kernel<<<M, 256, 0, stream>>>(x1, ffada, mask, ffnormH, 0);

    // --- stage 5: H = gelu_tanh(ff_norm @ ff_w1 + ff_b1)   (M x 2048) ---
    gemm_wmma_kernel<1><<<dim3(D2 / 256, M / 128), 256, 0, stream>>>(
        ffnormH, fw1t, D, D2, ff_b1, nullptr, nullptr, nullptr, H, nullptr);

    // --- stage 6: out = x1 + gate_mlp * (H @ ff_w2 + ff_b2) ---
    gemm_wmma_kernel<3><<<dim3(D / 256, M / 128), 256, 0, stream>>>(
        H, fw2t, D2, D, ff_b2, x1, ffada, nullptr, nullptr, out);
}
